// Attention_33517924778029
// MI455X (gfx1250) — matrix operand order repaired and checked by reading, not compiled, not run on hardware
//
#include <hip/hip_runtime.h>
#include <hip/hip_bf16.h>

// ---------------------------------------------------------------------------
// Multi-head attention forward, CDNA5 (gfx1250), wave32 +
// v_wmma_f32_16x16x32_bf16 + global_load_async_to_lds_b128 (ASYNCcnt) +
// DPP16 row_xmask softmax reductions (no LDS traffic in the reduce).
//   B=4, T=2048, HID=1024, H=16, D=64
// Pipeline:
//   1) gemm<fp32 in, OUT_SPLIT>   : Qp  = split_heads(q@Wq)*D^-0.5   bf16 [bh][t][64]
//   2) gemm<fp32 in, OUT_SPLIT>   : Kp  = split_heads(k@Wk)          bf16 [bh][t][64]
//   3) gemm<fp32 in, OUT_SPLIT_T> : Vt  = split_heads(v@Wv)^T        bf16 [bh][64][t]
//   4) flash_attn_wmma            : AttnC = softmax(Qp Kp^T + bias) Vp  (K/V double-
//                                   buffered in LDS via async-to-LDS copies)
//   5) gemm<bf16 in, OUT_F32>     : out = AttnC @ Wo                 fp32
// ---------------------------------------------------------------------------

typedef __attribute__((ext_vector_type(16))) __bf16 v16bf;
typedef __attribute__((ext_vector_type(8)))  float  v8f;

union Frag {            // one WMMA 16-bf16 operand fragment per lane
    uint4 u[2];
    v16bf v;
};

#define TOK   8192      // B*T rows in projection GEMMs
#define HIDN  1024
#define TSEQ  2048
#define NH    16
#define DH    64

enum { OUT_SPLIT = 0, OUT_SPLIT_T = 1, OUT_F32 = 2 };

// Generic->LDS byte offset: low 32 bits of a generic pointer to __shared__
// data are the wave-relative LDS offset (flat aperture mapping, ISA 10.2).
__device__ __forceinline__ unsigned lds_off(const void* p) {
    return (unsigned)(uintptr_t)p;
}

// Async global->LDS 16-byte copy per lane (VGLOBAL GLOBAL_LOAD_ASYNC_TO_LDS_B128,
// tracked with ASYNCcnt). vdst = LDS byte address, vaddr = 64-bit global address.
__device__ __forceinline__ void async_copy_b128(unsigned lds, const void* gaddr) {
    asm volatile("global_load_async_to_lds_b128 %0, %1, off"
                 :: "v"(lds), "v"(gaddr) : "memory");
}
__device__ __forceinline__ void wait_async_le4() {
    asm volatile("s_wait_asynccnt 0x4" ::: "memory");
}
__device__ __forceinline__ void wait_async_0() {
    asm volatile("s_wait_asynccnt 0x0" ::: "memory");
}

// DPP16 ROW_XMASK lane swap (lane ^= M within each 16-lane row): pure VALU,
// co-executes with WMMA, replaces ds_bpermute-based __shfl_xor.
template<int M>
__device__ __forceinline__ float dpp_xor(float x) {
    int r = __builtin_amdgcn_update_dpp(0, __builtin_bit_cast(int, x),
                                        0x160 | M, 0xf, 0xf, true);
    return __builtin_bit_cast(float, r);
}
__device__ __forceinline__ float row_max16(float x) {
    x = fmaxf(x, dpp_xor<1>(x));
    x = fmaxf(x, dpp_xor<2>(x));
    x = fmaxf(x, dpp_xor<4>(x));
    x = fmaxf(x, dpp_xor<8>(x));
    return x;
}
__device__ __forceinline__ float row_sum16(float x) {
    x += dpp_xor<1>(x);
    x += dpp_xor<2>(x);
    x += dpp_xor<4>(x);
    x += dpp_xor<8>(x);
    return x;
}

// ---------------------------------------------------------------------------
// Tiled GEMM: [8192 x 1024] x [1024 x 1024], bf16 WMMA, f32 accumulate.
// Block 256 thr (8 waves), tile 128x128, K-step 32. Waves 2(M) x 4(N),
// each wave 64x32 via 4x2 WMMA 16x16 tiles.
// fp32 A converted to bf16 while staging into LDS; bf16 A staged with
// async-to-LDS copies. W converted + transposed into LDS [n][k] so B
// fragments are contiguous-K.
// ---------------------------------------------------------------------------
template<bool A_BF16, int OMODE>
__global__ __launch_bounds__(256) void gemm_bf16_wmma(const void* __restrict__ Aptr,
                                                      const float* __restrict__ W,
                                                      void* __restrict__ Out,
                                                      float scale)
{
    __shared__ __bf16 As[128 * 32];
    __shared__ __bf16 Bs[128 * 32];

    const int tid  = threadIdx.x;
    const int lane = tid & 31;
    const int wave = tid >> 5;
    const int wm   = wave & 1;               // wave M index (0..1)
    const int wn   = wave >> 1;              // wave N index (0..3)
    const int m0   = blockIdx.y * 128;
    const int n0   = blockIdx.x * 128;
    const int l16  = lane & 15;
    const int hi   = lane >> 4;              // 0 = lanes 0-15, 1 = lanes 16-31

    v8f acc[4][2] = {};

    for (int k0 = 0; k0 < HIDN; k0 += 32) {
        // ---- stage A tile (128x32) as bf16 ----
        if (A_BF16) {
            const __bf16* Ab = (const __bf16*)Aptr;
            const unsigned asb = lds_off(&As[0]);
#pragma unroll
            for (int i = 0; i < 2; ++i) {
                int q   = tid + i * 256;             // 512 chunks of 8 bf16
                int row = q >> 2;
                int c   = (q & 3) * 8;
                async_copy_b128(asb + (unsigned)(row * 32 + c) * 2,
                                Ab + (size_t)(m0 + row) * HIDN + k0 + c);
            }
        } else {
            const float* Af = (const float*)Aptr;
#pragma unroll
            for (int i = 0; i < 4; ++i) {
                int q   = tid + i * 256;             // 1024 chunks of 4 f32
                int row = q >> 3;
                int c   = (q & 7) * 4;
                float4 f = *(const float4*)(Af + (size_t)(m0 + row) * HIDN + k0 + c);
                __bf16* dst = &As[row * 32 + c];
                dst[0] = (__bf16)f.x; dst[1] = (__bf16)f.y;
                dst[2] = (__bf16)f.z; dst[3] = (__bf16)f.w;
            }
        }
        // ---- stage W tile (32x128) into LDS transposed: Bs[n][k] ----
#pragma unroll
        for (int i = 0; i < 4; ++i) {
            int q  = tid + i * 256;                  // 1024 chunks of 4 f32
            int kk = q >> 5;
            int c  = (q & 31) * 4;
            float4 f = *(const float4*)(W + (size_t)(k0 + kk) * HIDN + n0 + c);
            Bs[(c + 0) * 32 + kk] = (__bf16)f.x;
            Bs[(c + 1) * 32 + kk] = (__bf16)f.y;
            Bs[(c + 2) * 32 + kk] = (__bf16)f.z;
            Bs[(c + 3) * 32 + kk] = (__bf16)f.w;
        }
        if (A_BF16) wait_async_0();
        __syncthreads();

        // ---- fragments per ISA layout ----
        Frag a[4], b[2];
#pragma unroll
        for (int mt = 0; mt < 4; ++mt) {
            int row = wm * 64 + mt * 16 + l16;
            int k1  = hi * 8;                        // A: K {0..7,16..23}/{8..15,24..31}
            a[mt].u[0] = *(const uint4*)&As[row * 32 + k1];
            a[mt].u[1] = *(const uint4*)&As[row * 32 + k1 + 16];
        }
#pragma unroll
        for (int nt = 0; nt < 2; ++nt) {
            int col = wn * 32 + nt * 16 + l16;
            int k1  = hi * 8;                        // B: K {0..7,16..23}/{8..15,24..31}
            b[nt].u[0] = *(const uint4*)&Bs[col * 32 + k1];
            b[nt].u[1] = *(const uint4*)&Bs[col * 32 + k1 + 16];
        }
#pragma unroll
        for (int mt = 0; mt < 4; ++mt)
#pragma unroll
            for (int nt = 0; nt < 2; ++nt)
                acc[mt][nt] = __builtin_amdgcn_wmma_f32_16x16x32_bf16(
                    false, a[mt].v, false, b[nt].v, (short)0, acc[mt][nt], false, false);
        __syncthreads();
    }

    // ---- epilogue: C layout lane(l16)=col, rows r(+8 for hi half) ----
#pragma unroll
    for (int mt = 0; mt < 4; ++mt) {
#pragma unroll
        for (int nt = 0; nt < 2; ++nt) {
#pragma unroll
            for (int r = 0; r < 8; ++r) {
                int grow = m0 + wm * 64 + mt * 16 + r + hi * 8;   // token row (b*T+t)
                int gcol = n0 + wn * 32 + nt * 16 + l16;          // hidden col (h*64+d)
                float v = acc[mt][nt][r] * scale;
                if (OMODE == OUT_F32) {
                    ((float*)Out)[(size_t)grow * HIDN + gcol] = v;
                } else {
                    int bh = (grow >> 11) * NH + (gcol >> 6);
                    int t  = grow & (TSEQ - 1);
                    int d  = gcol & (DH - 1);
                    size_t idx = (OMODE == OUT_SPLIT)
                        ? ((size_t)bh * TSEQ + t) * DH + d
                        : ((size_t)bh * DH + d) * TSEQ + t;
                    ((__bf16*)Out)[idx] = (__bf16)v;
                }
            }
        }
    }
}

// ---------------------------------------------------------------------------
// Flash attention: grid (T/64 qtiles, B*H). 128 thr = 4 waves, one 16-row
// WMMA M-tile per wave. Key chunks of 32: K (32x64) and V^T (64x32) tiles
// are double-buffered in LDS via global_load_async_to_lds_b128 so all 4
// waves share one copy and the next chunk's DMA overlaps this chunk's
// WMMA + online softmax. Softmax row reductions use DPP16 row_xmask
// (VALU-only). P staged through LDS (C-layout -> A-layout).
// ---------------------------------------------------------------------------
__global__ __launch_bounds__(128) void flash_attn_wmma(const __bf16* __restrict__ Qp,
                                                       const __bf16* __restrict__ Kp,
                                                       const __bf16* __restrict__ Vt,
                                                       const int*    __restrict__ mask,
                                                       __bf16*       __restrict__ OutC)
{
    __shared__ __bf16 Kt[2][32 * 64];   // [buf][key][d]   4 KB per buffer
    __shared__ __bf16 Vs[2][64 * 32];   // [buf][d][key]   4 KB per buffer
    __shared__ __bf16 Ps[4][16 * 32];   // per-wave 16x32 P tile

    const int tid  = threadIdx.x;
    const int lane = tid & 31;
    const int w    = tid >> 5;
    const int l16  = lane & 15;
    const int hi   = lane >> 4;

    const int bh    = blockIdx.y;                 // 0..63
    const int qbase = blockIdx.x * 64 + w * 16;
    const int bidx  = bh >> 4;
    const int hidx  = bh & 15;
    const int* mrow = mask + (size_t)(bh & 3) * TSEQ;   // reference: bias row = bh % B

    const unsigned kldsb = lds_off(&Kt[0][0]);
    const unsigned vldsb = lds_off(&Vs[0][0]);

    // issue one stage of async copies: 4 x b128 per thread (2 for K, 2 for V)
    auto stage = [&](int jc) {
        const int buf = jc & 1;
        const int j   = jc * 32;
        const __bf16* ksrc = Kp + ((size_t)bh * TSEQ + j) * DH;   // contiguous 4 KB
        const unsigned kb = kldsb + (unsigned)buf * (32 * 64 * 2);
#pragma unroll
        for (int i = 0; i < 2; ++i) {
            int c = tid + i * 128;                // 256 chunks of 8 bf16
            async_copy_b128(kb + (unsigned)c * 16, ksrc + c * 8);
        }
        const unsigned vb = vldsb + (unsigned)buf * (64 * 32 * 2);
#pragma unroll
        for (int i = 0; i < 2; ++i) {
            int c = tid + i * 128;                // 64 rows x 4 chunks of 8 bf16
            int d = c >> 2, o = (c & 3) * 8;
            async_copy_b128(vb + (unsigned)(d * 32 + o) * 2,
                            Vt + ((size_t)bh * DH + d) * TSEQ + j + o);
        }
    };

    // Q fragments (16 rows x 64 depth = two K=32 A-fragments), registers
    Frag aq[2];
#pragma unroll
    for (int kk = 0; kk < 2; ++kk) {
        const __bf16* p = Qp + ((size_t)bh * TSEQ + qbase + l16) * DH + kk * 32 + hi * 8;
        aq[kk].u[0] = *(const uint4*)p;
        aq[kk].u[1] = *(const uint4*)(p + 16);
    }

    v8f O[4] = {};
    float m_i[8], l_i[8];
#pragma unroll
    for (int r = 0; r < 8; ++r) { m_i[r] = -3.0e38f; l_i[r] = 0.0f; }

    const int NCH = TSEQ / 32;                    // 64 chunks
    stage(0);

    for (int jc = 0; jc < NCH; ++jc) {
        const int buf = jc & 1;
        const int j   = jc * 32;
        if (jc + 1 < NCH) {
            stage(jc + 1);        // prefetch next buffer
            wait_async_le4();     // oldest 4 (current buffer) complete, in order
        } else {
            wait_async_0();
        }
        __syncthreads();

        const __bf16* KB = &Kt[buf][0];
        const __bf16* VB = &Vs[buf][0];

        // ---- K B-fragments from LDS: 2 key tiles x 2 depth halves ----
        Frag bk[2][2];
#pragma unroll
        for (int kt = 0; kt < 2; ++kt)
#pragma unroll
            for (int kk = 0; kk < 2; ++kk) {
                const __bf16* p = KB + (kt * 16 + l16) * DH + kk * 32 + hi * 8;
                bk[kt][kk].u[0] = *(const uint4*)p;
                bk[kt][kk].u[1] = *(const uint4*)(p + 16);
            }

        // ---- S = Q K^T (16x32 over 2 N-tiles) ----
        v8f s[2];
#pragma unroll
        for (int kt = 0; kt < 2; ++kt) {
            v8f z = {};
            z = __builtin_amdgcn_wmma_f32_16x16x32_bf16(false, aq[0].v, false, bk[kt][0].v,
                                                        (short)0, z, false, false);
            s[kt] = __builtin_amdgcn_wmma_f32_16x16x32_bf16(false, aq[1].v, false, bk[kt][1].v,
                                                            (short)0, z, false, false);
        }
        float bias0 = (float)mrow[j + l16]      * -1e9f;
        float bias1 = (float)mrow[j + 16 + l16] * -1e9f;

        // ---- online softmax per row (rows in fixed half; cols across 16 lanes)
        float alpha[8];
#pragma unroll
        for (int r = 0; r < 8; ++r) {
            float s0 = s[0][r] + bias0;
            float s1 = s[1][r] + bias1;
            float cm = row_max16(fmaxf(s0, s1));            // DPP16 row_xmask
            float mn = fmaxf(m_i[r], cm);
            float al = __expf(m_i[r] - mn);
            float p0 = __expf(s0 - mn);
            float p1 = __expf(s1 - mn);
            float rs = row_sum16(p0 + p1);                  // DPP16 row_xmask
            l_i[r] = l_i[r] * al + rs;
            m_i[r] = mn;
            alpha[r] = al;
            int row = r + hi * 8;                           // C layout -> LDS
            Ps[w][row * 32 + l16]      = (__bf16)p0;
            Ps[w][row * 32 + 16 + l16] = (__bf16)p1;
        }
        __syncthreads();

        // ---- reload P as A-fragment (16x32) ----
        Frag ap;
        {
            int k1 = hi * 8;
            ap.u[0] = *(const uint4*)&Ps[w][l16 * 32 + k1];
            ap.u[1] = *(const uint4*)&Ps[w][l16 * 32 + k1 + 16];
        }

        // ---- V B-fragments from LDS [d][key] (contiguous in K) ----
        Frag bv[4];
#pragma unroll
        for (int dt = 0; dt < 4; ++dt) {
            const __bf16* p = VB + (dt * 16 + l16) * 32 + hi * 8;
            bv[dt].u[0] = *(const uint4*)p;
            bv[dt].u[1] = *(const uint4*)(p + 16);
        }

        // ---- rescale + accumulate O ----
#pragma unroll
        for (int dt = 0; dt < 4; ++dt) {
#pragma unroll
            for (int r = 0; r < 8; ++r) O[dt][r] *= alpha[r];
            O[dt] = __builtin_amdgcn_wmma_f32_16x16x32_bf16(false, ap.v, false, bv[dt].v,
                                                            (short)0, O[dt], false, false);
        }
        __syncthreads();   // buf + Ps safe to overwrite next chunk
    }

    // ---- normalize and store combined-head output (bf16) ----
#pragma unroll
    for (int dt = 0; dt < 4; ++dt) {
#pragma unroll
        for (int r = 0; r < 8; ++r) {
            int t   = qbase + r + hi * 8;
            int col = hidx * DH + dt * 16 + l16;
            float v = O[dt][r] * (1.0f / l_i[r]);
            OutC[((size_t)(bidx * TSEQ + t)) * HIDN + col] = (__bf16)v;
        }
    }
}

// ---------------------------------------------------------------------------
extern "C" void kernel_launch(void* const* d_in, const int* in_sizes, int n_in,
                              void* d_out, int out_size, void* d_ws, size_t ws_size,
                              hipStream_t stream)
{
    (void)in_sizes; (void)n_in; (void)out_size; (void)ws_size;
    const float* q    = (const float*)d_in[0];
    const float* k    = (const float*)d_in[1];
    const float* v    = (const float*)d_in[2];
    const int*   mask = (const int*)  d_in[3];
    const float* Wq   = (const float*)d_in[4];
    const float* Wk   = (const float*)d_in[5];
    const float* Wv   = (const float*)d_in[6];
    const float* Wo   = (const float*)d_in[7];

    const size_t NTOT = (size_t)TOK * HIDN;     // 8192*1024 elements
    __bf16* Qp    = (__bf16*)d_ws;
    __bf16* Kp    = Qp + NTOT;
    __bf16* Vt    = Kp + NTOT;
    __bf16* AttnC = Vt + NTOT;                  // 4 x 16 MB bf16 scratch

    dim3 gG(HIDN / 128, TOK / 128);             // (8, 64)
    gemm_bf16_wmma<false, OUT_SPLIT  ><<<gG, 256, 0, stream>>>(q, Wq, Qp, 0.125f); // D^-0.5
    gemm_bf16_wmma<false, OUT_SPLIT  ><<<gG, 256, 0, stream>>>(k, Wk, Kp, 1.0f);
    gemm_bf16_wmma<false, OUT_SPLIT_T><<<gG, 256, 0, stream>>>(v, Wv, Vt, 1.0f);

    flash_attn_wmma<<<dim3(TSEQ / 64, NH * 4), 128, 0, stream>>>(Qp, Kp, Vt, mask, AttnC);

    gemm_bf16_wmma<true, OUT_F32><<<gG, 256, 0, stream>>>(AttnC, Wo, d_out, 1.0f);
}
